// KANModel_35098472743379
// MI455X (gfx1250) — matrix operand order repaired and checked by reading, not compiled, not run on hardware
//
#include <hip/hip_runtime.h>

typedef __attribute__((ext_vector_type(16))) _Float16 v16h;
typedef __attribute__((ext_vector_type(8)))  float    v8f;

#define G_INT  5
#define KORD   3
#define NBAS   8            // G + K bases per activation
#define NFEAT  9            // silu + 8 bases
#define IN0    128
#define OUT0   64
#define K0     (IN0 * NFEAT)   // 1152
#define IN1    64
#define SA_STRIDE 1160      // padded f16 row stride (breaks 64-bank alignment)

// Cox-de-Boor cubic bases on grid extended by KORD uniform steps each side.
// Matches pykan B_batch(extend=True): interior knots are the original grid
// points, extension uses h = (g5-g0)/G.
__device__ __forceinline__ void spline_basis8(float x, const float* g6, float* out) {
    const float h = (g6[5] - g6[0]) * (1.0f / (float)G_INT);
    float t[12];
    t[0] = g6[0] - 3.0f * h;
    t[1] = g6[0] - 2.0f * h;
    t[2] = g6[0] - h;
#pragma unroll
    for (int p = 0; p < 6; ++p) t[3 + p] = g6[p];
    t[9]  = g6[5] + h;
    t[10] = g6[5] + 2.0f * h;
    t[11] = g6[5] + 3.0f * h;

    float Bv[11];
#pragma unroll
    for (int j = 0; j < 11; ++j)
        Bv[j] = (x >= t[j] && x < t[j + 1]) ? 1.0f : 0.0f;
#pragma unroll
    for (int kk = 1; kk <= KORD; ++kk) {
#pragma unroll
        for (int j = 0; j < 11 - kk; ++j) {
            float num0 = (x - t[j]) * Bv[j];
            float den0 = t[j + kk] - t[j];
            float num1 = (t[j + kk + 1] - x) * Bv[j + 1];
            float den1 = t[j + kk + 1] - t[j + 1];
            Bv[j] = num0 / den0 + num1 / den1;
        }
    }
#pragma unroll
    for (int j = 0; j < NBAS; ++j) out[j] = Bv[j];
}

__device__ __forceinline__ float silu_f(float x) {
    return x / (1.0f + __expf(-x));
}

// Pack layer-0 weights: W[o][i*9+0] = sb, W[o][i*9+1+c] = ssp*coef[c], as f16.
__global__ void kan_pack_w0(const float* __restrict__ coef0,
                            const float* __restrict__ sb0,
                            const float* __restrict__ ssp0,
                            _Float16* __restrict__ Wp0) {
    int idx = blockIdx.x * blockDim.x + threadIdx.x;
    if (idx >= OUT0 * IN0) return;
    int o = idx / IN0;
    int i = idx - o * IN0;
    _Float16* dst = Wp0 + (long)o * K0 + i * NFEAT;
    dst[0] = (_Float16)sb0[idx];
    float sp = ssp0[idx];
#pragma unroll
    for (int c = 0; c < NBAS; ++c)
        dst[1 + c] = (_Float16)(sp * coef0[(long)idx * NBAS + c]);
}

// Layer 0: gather embeddings -> f16 feature tile in LDS -> WMMA GEMM vs Wp0.
// One block = 16 batch rows; 4 waves each own a 16-wide output tile.
__global__ void __launch_bounds__(128) kan_layer0(
        const int* __restrict__ uidx, const int* __restrict__ iidx,
        const float* __restrict__ emb_user, const float* __restrict__ emb_item,
        const float* __restrict__ grid0, const float* __restrict__ bias0,
        const _Float16* __restrict__ Wp0, float* __restrict__ hbuf) {
    __shared__ _Float16 sA[16 * SA_STRIDE];
    __shared__ int s_u[16], s_i[16];
    const int t = threadIdx.x;
    const int mtile = blockIdx.x * 16;

    if (t < 16) { s_u[t] = uidx[mtile + t]; s_i[t] = iidx[mtile + t]; }
    __syncthreads();

    // --- feature expansion: thread t handles input column i = t for all 16 rows
    {
        const int i = t;
        float g6[6];
#pragma unroll
        for (int p = 0; p < 6; ++p) g6[p] = grid0[i * 6 + p];
        const float* src = (i < 64) ? emb_user : emb_item;
        const int col = (i < 64) ? i : (i - 64);
#pragma unroll 4
        for (int s = 0; s < 16; ++s) {
            long row = (long)((i < 64) ? s_u[s] : s_i[s]);
            float x = src[row * 64 + col];
            float f[NFEAT];
            f[0] = silu_f(x);
            spline_basis8(x, g6, f + 1);
            _Float16* dst = &sA[s * SA_STRIDE + i * NFEAT];
#pragma unroll
            for (int j = 0; j < NFEAT; ++j) dst[j] = (_Float16)f[j];
        }
    }
    __syncthreads();

    // --- WMMA: D(16x64) += A(16x1152) * B(1152x64), 36 k-steps of 32
    const int lane   = t & 31;
    const int wave   = t >> 5;
    const int ntile  = wave * 16;
    const int m      = lane & 15;
    const int n      = ntile + (lane & 15);
    const int khalfA = (lane < 16) ? 0 : 16;   // A: K striped 0-15 / 16-31
    const int khalfB = (lane < 16) ? 0 : 16;   // B: K striped 0-15 / 16-31

    v8f acc = {};
#pragma unroll 2
    for (int kt = 0; kt < K0 / 32; ++kt) {
        const int kb = kt * 32;
        const _Float16* arow = &sA[m * SA_STRIDE + kb + khalfA];
        v16h a;
#pragma unroll
        for (int p = 0; p < 4; ++p) {            // VGPRs 0-3: K = khalf+{0..7}
            a[2 * p]     = arow[2 * p];          // VGPRs 4-7: K = khalf+8+{0..7}
            a[2 * p + 1] = arow[2 * p + 1];
            a[2 * p + 8] = arow[8 + 2 * p];
            a[2 * p + 9] = arow[8 + 2 * p + 1];
        }
        const _Float16* brow = Wp0 + (long)n * K0 + kb + khalfB;
        v16h b;
#pragma unroll
        for (int p = 0; p < 16; ++p) b[p] = brow[p];
        if (kt + 1 < K0 / 32) __builtin_prefetch(brow + 32, 0, 0);

        acc = __builtin_amdgcn_wmma_f32_16x16x32_f16(
            /*neg_a=*/false, a, /*neg_b=*/false, b,
            /*c_mod=*/(short)0, acc, /*reuse_a=*/false, /*reuse_b=*/false);
    }

    const float bn = bias0[n];
#pragma unroll
    for (int r = 0; r < 8; ++r) {                // D layout: VGPR r -> M=r / r+8
        int mm = r + ((lane < 16) ? 0 : 8);
        hbuf[(long)(mtile + mm) * OUT0 + n] = acc[r] + bn;
    }
}

// Layer 1 (64 -> 1) + sigmoid: one thread per batch element, pure f32.
__global__ void kan_layer1(const float* __restrict__ hbuf,
                           const float* __restrict__ grid1,
                           const float* __restrict__ coef1,
                           const float* __restrict__ sb1,
                           const float* __restrict__ ssp1,
                           const float* __restrict__ bias1,
                           float* __restrict__ out, int B) {
    int b = blockIdx.x * blockDim.x + threadIdx.x;
    if (b >= B) return;
    float acc = bias1[0];
    for (int i = 0; i < IN1; ++i) {
        float x = hbuf[(long)b * OUT0 + i];
        float g6[6];
#pragma unroll
        for (int p = 0; p < 6; ++p) g6[p] = grid1[i * 6 + p];
        float bs[NBAS];
        spline_basis8(x, g6, bs);
        float s = 0.0f;
#pragma unroll
        for (int c = 0; c < NBAS; ++c) s += coef1[i * NBAS + c] * bs[c];
        acc += sb1[i] * silu_f(x) + ssp1[i] * s;
    }
    out[b] = 1.0f / (1.0f + __expf(-acc));
}

extern "C" void kernel_launch(void* const* d_in, const int* in_sizes, int n_in,
                              void* d_out, int out_size, void* d_ws, size_t ws_size,
                              hipStream_t stream) {
    (void)n_in; (void)out_size; (void)ws_size;
    const int*   uidx     = (const int*)  d_in[0];
    const int*   iidx     = (const int*)  d_in[1];
    // d_in[2], d_in[3]: grid_update_num / stop_grid_update_step (no effect on fwd)
    const float* emb_user = (const float*)d_in[4];
    const float* emb_item = (const float*)d_in[5];
    const float* grid0    = (const float*)d_in[6];
    const float* coef0    = (const float*)d_in[7];
    const float* sb0      = (const float*)d_in[8];
    const float* ssp0     = (const float*)d_in[9];
    const float* bias0    = (const float*)d_in[10];
    const float* grid1    = (const float*)d_in[11];
    const float* coef1    = (const float*)d_in[12];
    const float* sb1      = (const float*)d_in[13];
    const float* ssp1     = (const float*)d_in[14];
    const float* bias1    = (const float*)d_in[15];
    float* out = (float*)d_out;

    const int B = in_sizes[0];   // 1024 in setup; assumed multiple of 16

    // workspace layout: [ Wp0 f16 64x1152 | hbuf f32 Bx64 ]
    _Float16* Wp0  = (_Float16*)d_ws;
    float*    hbuf = (float*)((char*)d_ws + (size_t)OUT0 * K0 * sizeof(_Float16));

    kan_pack_w0<<<(OUT0 * IN0 + 255) / 256, 256, 0, stream>>>(coef0, sb0, ssp0, Wp0);
    kan_layer0<<<B / 16, 128, 0, stream>>>(uidx, iidx, emb_user, emb_item,
                                           grid0, bias0, Wp0, hbuf);
    kan_layer1<<<(B + 127) / 128, 128, 0, stream>>>(hbuf, grid1, coef1, sb1, ssp1,
                                                    bias1, out, B);
}
